// HardNegativeInfoNCELoss_11218454577735
// MI455X (gfx1250) — matrix operand order repaired and checked by reading, not compiled, not run on hardware
//
#include <hip/hip_runtime.h>
#include <hip/hip_bf16.h>
#include <math.h>

// ---------------------------------------------------------------------------
// HardNegativeInfoNCELoss for MI455X (gfx1250, wave32, WMMA bf16)
//   S = Q A^T / T via v_wmma_f32_16x16x32_bf16; 16x64 register tile per wave
//   (A-frag reused across 4 N-tiles, all B-frags resident so loads pipeline);
//   streaming online softmax -> the 256MB S matrix is never materialized.
//   Diagonal handled by a separate wave-per-row dot kernel (no divergence in
//   the GEMM hot loop).
// ---------------------------------------------------------------------------

typedef __attribute__((ext_vector_type(16))) __bf16 v16bf;
typedef __attribute__((ext_vector_type(8)))  __bf16 v8bf;
typedef __attribute__((ext_vector_type(8)))  float  v8f;

#define BATCH   8192
#define DIM     1024
#define NQID    1024
#define NCHUNK  8                       // N split into 8 chunks of 1024 cols
#define CHUNK_COLS (BATCH / NCHUNK)     // 1024
#define MBLOCK  128                     // rows per workgroup (8 waves x 16)
#define NT      4                       // N-tiles per wave pass (16x64 C tile)
#define INV_T   10.0f                   // 1 / TEMPERATURE

// ---- workspace layout (bytes) ---------------------------------------------
static constexpr size_t SZ_EMB   = (size_t)BATCH * DIM * 2;      // 16 MB each
static constexpr size_t OFF_QB   = 0;
static constexpr size_t OFF_AB   = OFF_QB + SZ_EMB;
static constexpr size_t SZ_P     = (size_t)BATCH * NCHUNK * 4;   // 256 KB each
static constexpr size_t OFF_PMAX = OFF_AB + SZ_EMB;
static constexpr size_t OFF_PSUM = OFF_PMAX + SZ_P;
static constexpr size_t OFF_PHN  = OFF_PSUM + SZ_P;
static constexpr size_t OFF_PIDX = OFF_PHN + SZ_P;
static constexpr size_t OFF_PDIAG= OFF_PIDX + SZ_P;
static constexpr size_t OFF_HIST = OFF_PDIAG + (size_t)BATCH * 4;
static constexpr size_t OFF_PART = OFF_HIST + (size_t)NQID * 4;
static constexpr int    RF_BLOCKS = BATCH / 256;                 // 32

__device__ __forceinline__ unsigned short f2bf_rne(float f) {
    unsigned u = __float_as_uint(f);
    unsigned r = u + 0x7fffu + ((u >> 16) & 1u);
    return (unsigned short)(r >> 16);
}

// ---- kernel 0: zero histogram ---------------------------------------------
__global__ void init_kernel(int* __restrict__ hist) {
    int i = blockIdx.x * blockDim.x + threadIdx.x;
    if (i < NQID) hist[i] = 0;
}

// ---- kernel 1: fp32 -> bf16 for Q and A (4 elems/thread) ------------------
__global__ void cvt_kernel(const float* __restrict__ q, const float* __restrict__ a,
                           unsigned int* __restrict__ qb, unsigned int* __restrict__ ab) {
    size_t i = (size_t)blockIdx.x * blockDim.x + threadIdx.x;   // index of float4
    size_t i4 = i * 4;
    float4 fq = *(const float4*)(q + i4);
    float4 fa = *(const float4*)(a + i4);
    unsigned int q0 = (unsigned)f2bf_rne(fq.x) | ((unsigned)f2bf_rne(fq.y) << 16);
    unsigned int q1 = (unsigned)f2bf_rne(fq.z) | ((unsigned)f2bf_rne(fq.w) << 16);
    unsigned int a0 = (unsigned)f2bf_rne(fa.x) | ((unsigned)f2bf_rne(fa.y) << 16);
    unsigned int a1 = (unsigned)f2bf_rne(fa.z) | ((unsigned)f2bf_rne(fa.w) << 16);
    qb[i * 2 + 0] = q0; qb[i * 2 + 1] = q1;
    ab[i * 2 + 0] = a0; ab[i * 2 + 1] = a1;
}

// ---- kernel 2: qid histogram ----------------------------------------------
__global__ void hist_kernel(const int* __restrict__ qid, int* __restrict__ hist) {
    int i = blockIdx.x * blockDim.x + threadIdx.x;
    if (i < BATCH) atomicAdd(&hist[qid[i] & (NQID - 1)], 1);
}

// ---- kernel 2b: diagonal dot products (one wave per row) ------------------
//   pdiag[i] = (1/T) * dot(Qb[i,:], Ab[i,:])   -- bf16 inputs, f32 accumulate
__global__ __launch_bounds__(256)
void diag_kernel(const unsigned short* __restrict__ Qb,
                 const unsigned short* __restrict__ Ab,
                 float* __restrict__ pdiag) {
    const int wave = threadIdx.x >> 5;
    const int lane = threadIdx.x & 31;
    const int row  = blockIdx.x * 8 + wave;
    const unsigned short* qp = Qb + (size_t)row * DIM + lane * (DIM / 32);
    const unsigned short* ap = Ab + (size_t)row * DIM + lane * (DIM / 32);
    float acc = 0.f;
#pragma unroll
    for (int k = 0; k < DIM / 32; k += 8) {             // 32 elems/lane, v8 chunks
        v8bf qv = *(const v8bf*)(qp + k);
        v8bf av = *(const v8bf*)(ap + k);
#pragma unroll
        for (int e = 0; e < 8; ++e) acc = fmaf((float)qv[e], (float)av[e], acc);
    }
#pragma unroll
    for (int off = 16; off > 0; off >>= 1) acc += __shfl_xor(acc, off, 32);
    if (lane == 0) pdiag[row] = acc * INV_T;
}

// ---- kernel 3: WMMA GEMM + streaming softmax stats ------------------------
// grid = (NCHUNK, BATCH/MBLOCK), block = 256 (8 waves); wave owns 16 rows and
// sweeps its chunk 64 columns (4 WMMA N-tiles) at a time.
__global__ __launch_bounds__(256)
void gemm_stats_kernel(const unsigned short* __restrict__ Qb,
                       const unsigned short* __restrict__ Ab,
                       const int*  __restrict__ qid,
                       float* __restrict__ pmax, float* __restrict__ psum,
                       float* __restrict__ phn,  int*   __restrict__ pidx) {
    const int chunk  = blockIdx.x;
    const int n_base = chunk * CHUNK_COLS;
    const int wave   = threadIdx.x >> 5;
    const int lane   = threadIdx.x & 31;
    const int half   = lane >> 4;        // K-half select for A/B frags
    const int l16    = lane & 15;
    const int mbase  = blockIdx.y * MBLOCK + wave * 16;

    // A-fragment source row for this lane (ISA: A lane row = lane&15)
    const unsigned short* qrow_ptr = Qb + (size_t)(mbase + l16) * DIM;

    // C-fragment row qids: C row = r + half*8
    int qrow[8];
#pragma unroll
    for (int r = 0; r < 8; ++r) qrow[r] = qid[mbase + half * 8 + r];

    float mval[8], ssum[8], hn[8];
    int   idx[8];
#pragma unroll
    for (int r = 0; r < 8; ++r) {
        mval[r] = -INFINITY; ssum[r] = 0.f; hn[r] = 0.f; idx[r] = 0;
    }

    for (int t = 0; t < CHUNK_COLS / (16 * NT); ++t) {
        const int n0 = n_base + t * 16 * NT;
        int bcol[NT];
        const unsigned short* bptr[NT];
#pragma unroll
        for (int j = 0; j < NT; ++j) {
            bcol[j] = n0 + j * 16 + l16;
            bptr[j] = Ab + (size_t)bcol[j] * DIM + half * 16;
            if (t + 1 < CHUNK_COLS / (16 * NT))          // prefetch next pass's B
                __builtin_prefetch(bptr[j] + (size_t)16 * NT * DIM, 0, 0);
        }

        v8f c[NT];
#pragma unroll
        for (int j = 0; j < NT; ++j) c[j] = (v8f){};

#pragma unroll 2
        for (int kk = 0; kk < DIM; kk += 32) {
            // A frag: elems 0..15 = K[kk+half*16 ..+16]
            v8bf alo = *(const v8bf*)(qrow_ptr + kk + half * 16);
            v8bf ahi = *(const v8bf*)(qrow_ptr + kk + half * 16 + 8);
            v16bf a;
#pragma unroll
            for (int e = 0; e < 8; ++e) { a[e] = alo[e]; a[8 + e] = ahi[e]; }
            // Load ALL B-frags into distinct regs first so the 8 b128 loads
            // clause together and the WMMAs can wait progressively.
            v16bf b0 = *(const v16bf*)(bptr[0] + kk);
            v16bf b1 = *(const v16bf*)(bptr[1] + kk);
            v16bf b2 = *(const v16bf*)(bptr[2] + kk);
            v16bf b3 = *(const v16bf*)(bptr[3] + kk);
            c[0] = __builtin_amdgcn_wmma_f32_16x16x32_bf16(
                       false, a, false, b0, (short)0, c[0], false, false);
            c[1] = __builtin_amdgcn_wmma_f32_16x16x32_bf16(
                       false, a, false, b1, (short)0, c[1], false, false);
            c[2] = __builtin_amdgcn_wmma_f32_16x16x32_bf16(
                       false, a, false, b2, (short)0, c[2], false, false);
            c[3] = __builtin_amdgcn_wmma_f32_16x16x32_bf16(
                       false, a, false, b3, (short)0, c[3], false, false);
        }

        int qcol[NT];
#pragma unroll
        for (int j = 0; j < NT; ++j) qcol[j] = qid[bcol[j]];

#pragma unroll
        for (int j = 0; j < NT; ++j) {
#pragma unroll
            for (int r = 0; r < 8; ++r) {
                const int   mrow  = mbase + half * 8 + r;    // C row for this VGPR
                const float s     = c[j][r] * INV_T;
                // hard negatives: same qid, off-diagonal. Rare (~1.5% of
                // groups) -> uniform scalar branch, EXEC untouched.
                const bool match = (qcol[j] == qrow[r]) && (bcol[j] != mrow);
                if (__any(match)) hn[r] += match ? __expf(s) : 0.f;
                // branchless online max / sum / first-argmax
                const float mn = fmaxf(mval[r], s);
                ssum[r] = ssum[r] * __expf(mval[r] - mn) + __expf(s - mn);
                idx[r]  = (s > mval[r]) ? bcol[j] : idx[r];
                mval[r] = mn;
            }
        }
    }

    // reduce across the 16 lanes that share each row (xor stays in halves)
#pragma unroll
    for (int r = 0; r < 8; ++r) {
        float m = mval[r], s = ssum[r], h = hn[r];
        int   ix = idx[r];
#pragma unroll
        for (int off = 1; off < 16; off <<= 1) {
            float om = __shfl_xor(m, off, 32);
            float os = __shfl_xor(s, off, 32);
            float oh = __shfl_xor(h, off, 32);
            int   oi = __shfl_xor(ix, off, 32);
            float mn = fmaxf(m, om);
            s = s * __expf(m - mn) + os * __expf(om - mn);
            ix = (om > m || (om == m && oi < ix)) ? oi : ix;
            m = mn; h += oh;
        }
        if (l16 == 0) {
            int    row = mbase + half * 8 + r;
            size_t o   = (size_t)row * NCHUNK + chunk;
            pmax[o] = m; psum[o] = s; phn[o] = h; pidx[o] = ix;
        }
    }
}

// ---- kernel 4: per-row finalize + deterministic block reduction -----------
__global__ __launch_bounds__(256)
void rowfinal_kernel(const float* __restrict__ pmax, const float* __restrict__ psum,
                     const float* __restrict__ phn,  const int*   __restrict__ pidx,
                     const float* __restrict__ pdiag, float* __restrict__ partials) {
    const int row = blockIdx.x * blockDim.x + threadIdx.x;
    float M = -INFINITY, S = 0.f, H = 0.f, bestv = -INFINITY;
    int best = -1;
#pragma unroll
    for (int c = 0; c < NCHUNK; ++c) {
        size_t o = (size_t)row * NCHUNK + c;
        float m = pmax[o], s = psum[o];
        float mn = fmaxf(M, m);
        S = S * __expf(M - mn) + s * __expf(m - mn);
        M = mn; H += phn[o];
        best  = (m > bestv) ? pidx[o] : best;           // ascending c keeps first
        bestv = fmaxf(bestv, m);
    }
    float base = -(pdiag[row] - M - __logf(S));         // -logp[row,row]
    float corr = (best == row) ? 1.f : 0.f;

    __shared__ float sb[256], sh[256], sc[256];
    sb[threadIdx.x] = base; sh[threadIdx.x] = H; sc[threadIdx.x] = corr;
    __syncthreads();
    for (int s2 = 128; s2 > 0; s2 >>= 1) {
        if (threadIdx.x < s2) {
            sb[threadIdx.x] += sb[threadIdx.x + s2];
            sh[threadIdx.x] += sh[threadIdx.x + s2];
            sc[threadIdx.x] += sc[threadIdx.x + s2];
        }
        __syncthreads();
    }
    if (threadIdx.x == 0) {
        partials[blockIdx.x * 3 + 0] = sb[0];
        partials[blockIdx.x * 3 + 1] = sh[0];
        partials[blockIdx.x * 3 + 2] = sc[0];
    }
}

// ---- kernel 5: scalar finalize --------------------------------------------
__global__ void finalize_kernel(const float* __restrict__ partials,
                                const int* __restrict__ hist,
                                float* __restrict__ out) {
    if (threadIdx.x != 0 || blockIdx.x != 0) return;
    float b = 0.f, h = 0.f, c = 0.f;
    for (int i = 0; i < RF_BLOCKS; ++i) {
        b += partials[i * 3 + 0];
        h += partials[i * 3 + 1];
        c += partials[i * 3 + 2];
    }
    long long cnt = 0;
    for (int q = 0; q < NQID; ++q) {
        long long v = hist[q];
        cnt += v * (v - 1);
    }
    float total = b / (float)BATCH;
    if (cnt > 0) total += 1.0f /*HARD_NEG_WEIGHT*/ * (h / (float)cnt);
    out[0] = total;
    out[1] = c / (float)BATCH;
}

// ---------------------------------------------------------------------------
extern "C" void kernel_launch(void* const* d_in, const int* in_sizes, int n_in,
                              void* d_out, int out_size, void* d_ws, size_t ws_size,
                              hipStream_t stream) {
    const float* q   = (const float*)d_in[0];
    const float* a   = (const float*)d_in[1];
    const int*   qid = (const int*)d_in[2];
    float*       out = (float*)d_out;
    char*        ws  = (char*)d_ws;

    unsigned short* Qb    = (unsigned short*)(ws + OFF_QB);
    unsigned short* Ab    = (unsigned short*)(ws + OFF_AB);
    float*          pmax  = (float*)(ws + OFF_PMAX);
    float*          psum  = (float*)(ws + OFF_PSUM);
    float*          phn   = (float*)(ws + OFF_PHN);
    int*            pidx  = (int*)  (ws + OFF_PIDX);
    float*          pdiag = (float*)(ws + OFF_PDIAG);
    int*            hist  = (int*)  (ws + OFF_HIST);
    float*          part  = (float*)(ws + OFF_PART);

    init_kernel<<<NQID / 256, 256, 0, stream>>>(hist);

    const size_t n4 = (size_t)BATCH * DIM / 4;          // float4 chunks
    cvt_kernel<<<(unsigned)(n4 / 256), 256, 0, stream>>>(
        q, a, (unsigned int*)Qb, (unsigned int*)Ab);

    hist_kernel<<<BATCH / 256, 256, 0, stream>>>(qid, hist);

    diag_kernel<<<BATCH / 8, 256, 0, stream>>>(Qb, Ab, pdiag);

    dim3 grid(NCHUNK, BATCH / MBLOCK);                  // 8 x 64 workgroups
    gemm_stats_kernel<<<grid, 256, 0, stream>>>(Qb, Ab, qid,
                                                pmax, psum, phn, pidx);

    rowfinal_kernel<<<RF_BLOCKS, 256, 0, stream>>>(pmax, psum, phn, pidx, pdiag, part);

    finalize_kernel<<<1, 32, 0, stream>>>(part, hist, out);
}
